// mod_slg2_5282809774457
// MI455X (gfx1250) — compile-verified, matrix operand order checked by reading, not run on hardware
//
#include <hip/hip_runtime.h>
#include <hip/hip_bf16.h>
#include <hip/hip_fp16.h>

typedef _Float16 v16h __attribute__((ext_vector_type(16)));
typedef _Float16 v8h  __attribute__((ext_vector_type(8)));
typedef float    v8f  __attribute__((ext_vector_type(8)));

#define CDIV(a,b) (((a)+(b)-1)/(b))

// ---------------------------------------------------------------- helpers

__device__ __forceinline__ float geluf(float x) {
    return 0.5f * x * (1.0f + erff(x * 0.70710678118654752f));
}

__device__ __forceinline__ void atomicMaxF(float* a, float v) {
    if (v >= 0.0f) atomicMax((int*)a, __float_as_int(v));
    else           atomicMin((unsigned int*)a, __float_as_uint(v));
}
__device__ __forceinline__ void atomicMinF(float* a, float v) {
    if (v >= 0.0f) atomicMin((int*)a, __float_as_int(v));
    else           atomicMax((unsigned int*)a, __float_as_uint(v));
}

// Accumulator pre-loaded with per-column bias (col = lane&15), rows share it.
__device__ __forceinline__ v8f acc_init(const float* bias) {
    float b = bias ? bias[threadIdx.x & 15] : 0.0f;
    v8f a = {b, b, b, b, b, b, b, b};
    return a;
}

// WMMA inner loop: A row-major [16 x lda] tile in LDS, Bt column-major
// (Bt[n*ldb + k]) in LDS. KLEN compile-time (multiple of 32) -> full unroll.
// CDNA5 f16 A-frag: lane l -> row l&15, koff=(l>>4)*8; halves 0..7 = K koff..koff+7,
// halves 8..15 = K koff+16..+23. B-frag mirrored (col = l&15).
template <int KLEN>
__device__ __forceinline__ v8f wmma_loop(const _Float16* A, int lda,
                                         const _Float16* Bt, int ldb, v8f acc) {
    int lane = threadIdx.x & 31;
    int r  = lane & 15;
    int ko = (lane >> 4) << 3;
    const _Float16* pa = A  + r * lda + ko;
    const _Float16* pb = Bt + r * ldb + ko;
#pragma unroll
    for (int k = 0; k < KLEN; k += 32) {
        v8h a0 = *(const v8h*)(pa + k);
        v8h a1 = *(const v8h*)(pa + k + 16);
        v8h b0 = *(const v8h*)(pb + k);
        v8h b1 = *(const v8h*)(pb + k + 16);
        v16h af = __builtin_shufflevector(a0, a1, 0,1,2,3,4,5,6,7,8,9,10,11,12,13,14,15);
        v16h bf = __builtin_shufflevector(b0, b1, 0,1,2,3,4,5,6,7,8,9,10,11,12,13,14,15);
        acc = __builtin_amdgcn_wmma_f32_16x16x32_f16(false, af, false, bf,
                                                     (short)0, acc, false, false);
    }
    return acc;
}

// Store 16x16 f32 tile: lane l, vgpr r -> row r + 8*(l>>4), col l&15.
// Fast unguarded path for full tiles (all but the last grid block).
__device__ __forceinline__ void store_tile(float* out, int ldo, v8f c, int rows_valid) {
    int lane = threadIdx.x & 31;
    int col = lane & 15;
    int rb  = (lane >> 4) << 3;
    if (rows_valid >= 16) {
#pragma unroll
        for (int r = 0; r < 8; ++r) out[(size_t)(rb + r) * ldo + col] = c[r];
    } else {
#pragma unroll
        for (int r = 0; r < 8; ++r) {
            int rr = rb + r;
            if (rr < rows_valid) out[(size_t)rr * ldo + col] = c[r];
        }
    }
}

// ---------------------------------------------------------------- tiny utility kernels

__global__ void k_fill(float* p, float v, int n) {
    int i = blockIdx.x * blockDim.x + threadIdx.x;
    if (i < n) p[i] = v;
}

__global__ void k_cvt_w(const float* W, _Float16* Wt, int K, int N) {
    int i = blockIdx.x * blockDim.x + threadIdx.x;
    if (i >= K * N) return;
    int k = i / N, n = i % N;
    Wt[n * K + k] = (_Float16)W[k * N + n];   // transpose to column-major-by-output
}

__global__ void k_deg(const int* dst, int E, float* deg) {
    int i = blockIdx.x * blockDim.x + threadIdx.x;
    if (i < E) atomicAdd(&deg[dst[i]], 1.0f);
}

__global__ void k_rsqrt_ip(float* p, int n) {
    int i = blockIdx.x * blockDim.x + threadIdx.x;
    if (i < n) p[i] = rsqrtf(p[i]);
}

__global__ void k_l2batch(const int* e_idx, const int* ue, const int* ei0,
                          const int* batch, int* l2b, int L2) {
    int j = blockIdx.x * blockDim.x + threadIdx.x;
    if (j < L2) l2b[j] = batch[ei0[ue[e_idx[j]]]];
}

// ---------------------------------------------------------------- GCN pieces

// Out[i,c] = Hlin[i,c]*dinv[i]^2 + b[c]   (self-loop + bias)
__global__ void k_selfbias(float* Out, const float* Hl, const float* dinv,
                           const float* b, int n) {
    long i = (long)blockIdx.x * blockDim.x + threadIdx.x;
    if (i >= (long)n * 128) return;
    int r = (int)(i >> 7), c = (int)(i & 127);
    float di = dinv[r];
    Out[i] = Hl[i] * di * di + b[c];
}

// scatter: Out[dst] += Hlin[src] * dinv[src]*dinv[dst]; 32 threads per edge (4 cols each)
__global__ void k_scatter(float* Out, const float* Hl, const int* src, const int* dst,
                          const float* dinv, int E) {
    long gid = (long)blockIdx.x * blockDim.x + threadIdx.x;
    if (gid >= (long)E * 32) return;
    int e = (int)(gid >> 5);
    int q = (int)(gid & 31) * 4;
    int s = src[e], d = dst[e];
    float nrm = dinv[s] * dinv[d];
    float4 v = *(const float4*)(Hl + (size_t)s * 128 + q);
    float* o = Out + (size_t)d * 128 + q;
    atomicAdd(o + 0, v.x * nrm);
    atomicAdd(o + 1, v.y * nrm);
    atomicAdd(o + 2, v.z * nrm);
    atomicAdd(o + 3, v.w * nrm);
}

// ---------------------------------------------------------------- WMMA GEMM kernels

// Hlin = node_emb[x] @ W (K=128, N=128), 64 rows / block, 8 waves
__global__ void k_gemm_emb(const float* emb, const int* xid, const _Float16* Wt,
                           float* Out, int rows) {
    extern __shared__ _Float16 sm[];
    _Float16* A = sm;             // 64*128
    _Float16* W = sm + 64 * 128;  // 128*128
    __shared__ int sid[64];
    int row0 = blockIdx.x * 64, t = threadIdx.x;
    if (t < 64) { int r = row0 + t; sid[t] = (r < rows) ? xid[r] : -1; }
    for (int i = t; i < 128 * 128; i += 256) W[i] = Wt[i];
    __syncthreads();
    for (int i = t; i < 64 * 128; i += 256) {
        int r = i >> 7, c = i & 127;
        A[i] = (_Float16)((sid[r] >= 0) ? emb[sid[r] * 128 + c] : 0.0f);
    }
    __syncthreads();
    int wv = t >> 5;
    int valid = rows - row0; if (valid > 64) valid = 64;
    for (int j = wv; j < 32; j += 8) {
        int rt = j >> 3, nt = j & 7;
        v8f acc = acc_init(nullptr);
        acc = wmma_loop<128>(A + rt * 16 * 128, 128, W + nt * 16 * 128, 128, acc);
        store_tile(Out + (size_t)(row0 + rt * 16) * 128 + nt * 16, 128,
                   acc, valid - rt * 16);
    }
}

// Out = X @ W (X rows x 128 fp32, K=128, N=128)
__global__ void k_gemm_h(const float* X, const _Float16* Wt, float* Out, int rows) {
    extern __shared__ _Float16 sm[];
    _Float16* A = sm;
    _Float16* W = sm + 64 * 128;
    int row0 = blockIdx.x * 64, t = threadIdx.x;
    for (int i = t; i < 128 * 128; i += 256) W[i] = Wt[i];
    for (int i = t; i < 64 * 128; i += 256) {
        int r = i >> 7, c = i & 127;
        int rr = row0 + r;
        A[i] = (_Float16)((rr < rows) ? X[(size_t)rr * 128 + c] : 0.0f);
    }
    __syncthreads();
    int wv = t >> 5;
    int valid = rows - row0; if (valid > 64) valid = 64;
    for (int j = wv; j < 32; j += 8) {
        int rt = j >> 3, nt = j & 7;
        v8f acc = acc_init(nullptr);
        acc = wmma_loop<128>(A + rt * 16 * 128, 128, W + nt * 16 * 128, 128, acc);
        store_tile(Out + (size_t)(row0 + rt * 16) * 128 + nt * 16, 128,
                   acc, valid - rt * 16);
    }
}

// edge-FFN first GEMM. A tile = [H0[u] | edge_emb[attr] | H0[v]] (K=384) staged once;
// Z1 = c1@W0+b0, Z2 = c2@W0+b0 where c2 permutes the segment->W0-block pairing.
__global__ void k_ffn_ne_gemm(const float* H0, const float* edge_emb, const int* edge_attr,
                              const int* ue, const int* ei0, const int* ei1,
                              const _Float16* W0t, const float* b0,
                              float* Z1, float* Z2, int EU) {
    extern __shared__ _Float16 sm[];
    _Float16* A = sm;             // 64*384
    _Float16* W = sm + 64 * 384;  // 64*384 (Wt[n*384+k])
    __shared__ int su[64], sv[64], sa[64];
    int row0 = blockIdx.x * 64, t = threadIdx.x;
    if (t < 64) {
        int k = row0 + t;
        if (k < EU) { int e = ue[k]; su[t] = ei0[e]; sv[t] = ei1[e]; sa[t] = edge_attr[e]; }
        else su[t] = -1;
    }
    for (int i = t; i < 64 * 384; i += 256) W[i] = W0t[i];
    __syncthreads();
    // three power-of-2 segment loops (avoid /384, %384)
    for (int i = t; i < 64 * 128; i += 256) {
        int r = i >> 7, c = i & 127;
        A[r * 384 + c] =
            (_Float16)((su[r] >= 0) ? H0[(size_t)su[r] * 128 + c] : 0.0f);
    }
    for (int i = t; i < 64 * 128; i += 256) {
        int r = i >> 7, c = i & 127;
        A[r * 384 + 128 + c] =
            (_Float16)((su[r] >= 0) ? edge_emb[sa[r] * 128 + c] : 0.0f);
    }
    for (int i = t; i < 64 * 128; i += 256) {
        int r = i >> 7, c = i & 127;
        A[r * 384 + 256 + c] =
            (_Float16)((su[r] >= 0) ? H0[(size_t)sv[r] * 128 + c] : 0.0f);
    }
    __syncthreads();
    int wv = t >> 5;
    int z = wv >> 2, nt = wv & 3;     // waves 0-3 -> Z1, 4-7 -> Z2; nt = N tile
    int valid = EU - row0; if (valid > 64) valid = 64;
    for (int rt = 0; rt < 4; ++rt) {
        v8f acc = acc_init(b0 + nt * 16);
#pragma unroll
        for (int s = 0; s < 3; ++s) {
            int aseg = (z == 0) ? s * 128 : (2 - s) * 128;  // c2 = [V,E,U]
            acc = wmma_loop<128>(A + rt * 16 * 384 + aseg, 384,
                                 W + nt * 16 * 384 + s * 128, 384, acc);
        }
        float* Zo = z ? Z2 : Z1;
        store_tile(Zo + (size_t)(row0 + rt * 16) * 64 + nt * 16, 64,
                   acc, valid - rt * 16);
    }
}

// pair-FFN first GEMM: A = [h_edge[f] | h_edge[e]] (K=128), Z1/Z2 with half swap.
__global__ void k_ffn_ef_gemm(const float* h_edge, const int* e_idx, const int* f_idx,
                              const _Float16* W0t, const float* b0,
                              float* Z1, float* Z2, int L2) {
    extern __shared__ _Float16 sm[];
    _Float16* A = sm;             // 64*128
    _Float16* W = sm + 64 * 128;  // 128*128
    __shared__ int se[64], sf[64];
    int row0 = blockIdx.x * 64, t = threadIdx.x;
    if (t < 64) {
        int j = row0 + t;
        if (j < L2) { se[t] = e_idx[j]; sf[t] = f_idx[j]; }
        else se[t] = -1;
    }
    for (int i = t; i < 128 * 128; i += 256) W[i] = W0t[i];
    __syncthreads();
    for (int i = t; i < 64 * 64; i += 256) {
        int r = i >> 6, c = i & 63;
        A[(r << 7) + c] =
            (_Float16)((se[r] >= 0) ? h_edge[(size_t)sf[r] * 64 + c] : 0.0f);
    }
    for (int i = t; i < 64 * 64; i += 256) {
        int r = i >> 6, c = i & 63;
        A[(r << 7) + 64 + c] =
            (_Float16)((se[r] >= 0) ? h_edge[(size_t)se[r] * 64 + c] : 0.0f);
    }
    __syncthreads();
    int wv = t >> 5;
    int valid = L2 - row0; if (valid > 64) valid = 64;
    for (int j = wv; j < 64; j += 8) {
        int rt = j >> 4, combo = j & 15;
        int z = combo >> 3, nt = combo & 7;
        v8f acc = acc_init(b0 + nt * 16);
#pragma unroll
        for (int s = 0; s < 2; ++s) {
            int aseg = (z == 0) ? s * 64 : (1 - s) * 64;
            acc = wmma_loop<64>(A + rt * 16 * 128 + aseg, 128,
                                W + nt * 16 * 128 + s * 64, 128, acc);
        }
        float* Zo = z ? Z2 : Z1;
        store_tile(Zo + (size_t)(row0 + rt * 16) * 128 + nt * 16, 128,
                   acc, valid - rt * 16);
    }
}

// FFN second stage: G = 0.5*(gelu(bn1(Z1)) + gelu(bn2(Z2))); Out = G @ W1 + b1
template <int K, int N>
__global__ void k_ffn_part2(const float* Z1, const float* Z2,
                            const float* sc1, const float* sh1,
                            const float* sc2, const float* sh2,
                            const _Float16* W1t, const float* b1,
                            float* Out, int rows) {
    extern __shared__ _Float16 sm[];
    _Float16* Gm = sm;            // 64*K
    _Float16* W  = sm + 64 * K;   // N*K
    int row0 = blockIdx.x * 64, t = threadIdx.x;
    for (int i = t; i < N * K; i += 256) W[i] = W1t[i];
    for (int i = t; i < 64 * K; i += 256) {
        int r = i / K, c = i % K;      // K power of 2 -> shifts
        int rr = row0 + r;
        float v = 0.0f;
        if (rr < rows) {
            float z1 = Z1[(size_t)rr * K + c] * sc1[c] + sh1[c];
            float z2 = Z2[(size_t)rr * K + c] * sc2[c] + sh2[c];
            v = 0.5f * (geluf(z1) + geluf(z2));
        }
        Gm[i] = (_Float16)v;
    }
    __syncthreads();
    int wv = t >> 5;
    constexpr int NT = N >> 4;
    int valid = rows - row0; if (valid > 64) valid = 64;
    for (int j = wv; j < 4 * NT; j += 8) {
        int rt = j / NT, nt = j % NT;
        v8f acc = acc_init(b1 + nt * 16);
        acc = wmma_loop<K>(Gm + rt * 16 * K, K, W + nt * 16 * K, K, acc);
        store_tile(Out + (size_t)(row0 + rt * 16) * N + nt * 16, N,
                   acc, valid - rt * 16);
    }
}

// ---------------------------------------------------------------- batch-norm stats

#define CS_RPB 2048
template <int COLS>
__global__ void k_colstats(const float* Z, int rows, float* sum, float* sq) {
    __shared__ float ls[256], lq[256];
    int t = threadIdx.x;
    int col = t % COLS, rg = t / COLS;
    constexpr int NRG = 256 / COLS;
    int r0 = blockIdx.x * CS_RPB;
    int r1 = r0 + CS_RPB; if (r1 > rows) r1 = rows;
    float s = 0.0f, q = 0.0f;
    for (int r = r0 + rg; r < r1; r += NRG) {
        float v = Z[(size_t)r * COLS + col];
        s += v; q += v * v;
    }
    ls[t] = s; lq[t] = q;
    __syncthreads();
    if (rg == 0) {
#pragma unroll
        for (int g = 1; g < NRG; ++g) { s += ls[g * COLS + col]; q += lq[g * COLS + col]; }
        atomicAdd(&sum[col], s);
        atomicAdd(&sq[col], q);
    }
}

__global__ void k_bnfin(const float* sum, const float* sq, const float* g, const float* be,
                        float* sc, float* sh, int cols, float inv_n) {
    int c = blockIdx.x * blockDim.x + threadIdx.x;
    if (c >= cols) return;
    float m = sum[c] * inv_n;
    float v = sq[c] * inv_n - m * m;
    float s = g[c] * rsqrtf(v + 1e-5f);
    sc[c] = s;
    sh[c] = be[c] - m * s;
}

// ---------------------------------------------------------------- pooling

__global__ void k_pool_init(float* mn, float* mx, float* sm, float* c0, float* c1,
                            int n, int G) {
    int i = blockIdx.x * blockDim.x + threadIdx.x;
    if (i < n) { mn[i] = INFINITY; mx[i] = -INFINITY; sm[i] = 0.0f; }
    if (i < G) { c0[i] = 0.0f; c1[i] = 0.0f; }
}

__global__ void k_pool_acc(const float* X, const int* seg, int rows,
                           float* mn, float* mx, float* sm, float* cnt,
                           int relu, int addcnt) {
    long gid = (long)blockIdx.x * blockDim.x + threadIdx.x;
    if (gid >= (long)rows * 32) return;
    int r = (int)(gid >> 5);
    int q = (int)(gid & 31) * 4;
    int g = seg[r];
    float4 v = *(const float4*)(X + (size_t)r * 128 + q);
    float vv[4] = {v.x, v.y, v.z, v.w};
#pragma unroll
    for (int i = 0; i < 4; ++i) {
        float f = relu ? fmaxf(vv[i], 0.0f) : vv[i];
        int off = g * 128 + q + i;
        atomicMaxF(&mx[off], f);
        atomicMinF(&mn[off], f);
        atomicAdd(&sm[off], f);
    }
    if (addcnt && (gid & 31) == 0) atomicAdd(&cnt[g], 1.0f);
}

__global__ void k_pool_fin(const float* mn, const float* mx, const float* sm,
                           const float* c0, const float* c1, float* Hp, int G) {
    int i = blockIdx.x * blockDim.x + threadIdx.x;
    if (i >= G * 1152) return;
    int g = i / 1152, j = i % 1152;
    int s = j / 384, t = (j % 384) / 128, c = j % 128;
    float cnt = (s == 0) ? c0[g] : c1[g];
    int off = s * G * 128 + g * 128 + c;
    float v;
    if (t == 0)      v = (cnt > 0.0f) ? mn[off] : 0.0f;
    else if (t == 1) v = (cnt > 0.0f) ? mx[off] : 0.0f;
    else             v = sm[off] / fmaxf(cnt, 1.0f);
    Hp[i] = v;
}

// ---------------------------------------------------------------- head MLP (256 rows)

__device__ __forceinline__ float bsum128(float v, float* buf) {
    int t = threadIdx.x;
    __syncthreads();
    buf[t] = v;
    __syncthreads();
    for (int s = 64; s > 0; s >>= 1) { if (t < s) buf[t] += buf[t + s]; __syncthreads(); }
    return buf[0];
}

__global__ void k_head(const float* Hp,
                       const float* W1, const float* b1, const float* g1, const float* be1,
                       const float* W2, const float* b2, const float* g2, const float* be2,
                       const float* W3, const float* b3, float* out) {
    __shared__ float xr[1152];
    __shared__ float hb[128];
    __shared__ float red[128];
    int g = blockIdx.x, t = threadIdx.x;
    for (int i = t; i < 1152; i += 128) xr[i] = Hp[(size_t)g * 1152 + i];
    __syncthreads();
    float a = b1[t];
    for (int k = 0; k < 1152; ++k) a = fmaf(xr[k], W1[k * 128 + t], a);
    float s  = bsum128(a, red);
    float sq = bsum128(a * a, red);
    float m = s * (1.0f / 128.0f), var = sq * (1.0f / 128.0f) - m * m;
    float y = geluf((a - m) * rsqrtf(var + 1e-5f) * g1[t] + be1[t]);
    __syncthreads();
    hb[t] = y;
    __syncthreads();
    float a2 = b2[t];
    for (int k = 0; k < 128; ++k) a2 = fmaf(hb[k], W2[k * 128 + t], a2);
    s  = bsum128(a2, red);
    sq = bsum128(a2 * a2, red);
    m = s * (1.0f / 128.0f); var = sq * (1.0f / 128.0f) - m * m;
    float y2 = geluf((a2 - m) * rsqrtf(var + 1e-5f) * g2[t] + be2[t]);
    float o = bsum128(y2 * W3[t], red);
    if (t == 0) out[g] = o + b3[0];
}

// ---------------------------------------------------------------- host driver

extern "C" void kernel_launch(void* const* d_in, const int* in_sizes, int n_in,
                              void* d_out, int out_size, void* d_ws, size_t ws_size,
                              hipStream_t stream) {
    const int N   = in_sizes[0];
    const int E   = in_sizes[1];
    const int EU  = in_sizes[4];
    const int L2  = in_sizes[5] / 2;
    const int EL2 = in_sizes[6] / 2;
    const int G   = out_size;          // OUT == 1

    const int* x         = (const int*)d_in[0];
    const int* edge_attr = (const int*)d_in[1];
    const int* ei0       = (const int*)d_in[2];
    const int* ei1       = ei0 + E;
    const int* batch     = (const int*)d_in[3];
    const int* ue        = (const int*)d_in[4];
    const int* e_idx     = (const int*)d_in[5];
    const int* f_idx     = e_idx + L2;
    const int* l2e0      = (const int*)d_in[6];
    const int* l2e1      = l2e0 + EL2;

    const float* node_emb = (const float*)d_in[8];
    const float* edge_emb = (const float*)d_in[9];
    const float* gcn1_W = (const float*)d_in[10]; const float* gcn1_b = (const float*)d_in[11];
    const float* neW0 = (const float*)d_in[12]; const float* ne_b0 = (const float*)d_in[13];
    const float* ne_g0 = (const float*)d_in[14]; const float* ne_be0 = (const float*)d_in[15];
    const float* neW1 = (const float*)d_in[16]; const float* ne_b1 = (const float*)d_in[17];
    const float* efW0 = (const float*)d_in[18]; const float* ef_b0 = (const float*)d_in[19];
    const float* ef_g0 = (const float*)d_in[20]; const float* ef_be0 = (const float*)d_in[21];
    const float* efW1 = (const float*)d_in[22]; const float* ef_b1 = (const float*)d_in[23];
    const float* gcn2_W = (const float*)d_in[24]; const float* gcn2_b = (const float*)d_in[25];
    const float* hW1 = (const float*)d_in[26]; const float* hb1 = (const float*)d_in[27];
    const float* hg1 = (const float*)d_in[28]; const float* hbe1 = (const float*)d_in[29];
    const float* hW2 = (const float*)d_in[30]; const float* hb2 = (const float*)d_in[31];
    const float* hg2 = (const float*)d_in[32]; const float* hbe2 = (const float*)d_in[33];
    const float* hW3 = (const float*)d_in[34]; const float* hb3 = (const float*)d_in[35];

    // ---- workspace bump allocator (256B aligned)
    char* ws = (char*)d_ws;
    size_t off = 0;
    auto alloc = [&](size_t bytes) -> void* {
        off = (off + 255) & ~(size_t)255;
        void* p = ws + off;
        off += bytes;
        return p;
    };

    _Float16* gcn1Wt = (_Float16*)alloc(128 * 128 * 2);
    _Float16* neW0t  = (_Float16*)alloc(64 * 384 * 2);
    _Float16* neW1t  = (_Float16*)alloc(64 * 64 * 2);
    _Float16* efW0t  = (_Float16*)alloc(128 * 128 * 2);
    _Float16* efW1t  = (_Float16*)alloc(128 * 128 * 2);
    _Float16* gcn2Wt = (_Float16*)alloc(128 * 128 * 2);

    float* dinv1 = (float*)alloc((size_t)N * 4);
    float* dinv2 = (float*)alloc((size_t)L2 * 4);
    int*   l2b   = (int*)alloc((size_t)L2 * 4);

    float* Hlin  = (float*)alloc((size_t)N * 128 * 4);
    float* H0    = (float*)alloc((size_t)N * 128 * 4);
    float* Zne1  = (float*)alloc((size_t)EU * 64 * 4);
    float* Zne2  = (float*)alloc((size_t)EU * 64 * 4);
    float* h_edge= (float*)alloc((size_t)EU * 64 * 4);
    float* Zef1  = (float*)alloc((size_t)L2 * 128 * 4);
    float* Zef2  = (float*)alloc((size_t)L2 * 128 * 4);
    float* H1    = (float*)alloc((size_t)L2 * 128 * 4);
    float* Hlin2 = Zef1;   // reuse after Zef consumed
    float* H2    = Zef2;   // reuse after Zef consumed

    float* ne_stats = (float*)alloc(8 * 64 * 4);   // sum1,sq1,sum2,sq2,sc1,sh1,sc2,sh2
    float* ef_stats = (float*)alloc(8 * 128 * 4);
    float* ne_sum1 = ne_stats, *ne_sq1 = ne_stats+64, *ne_sum2 = ne_stats+128, *ne_sq2 = ne_stats+192;
    float* ne_sc1 = ne_stats+256, *ne_sh1 = ne_stats+320, *ne_sc2 = ne_stats+384, *ne_sh2 = ne_stats+448;
    float* ef_sum1 = ef_stats, *ef_sq1 = ef_stats+128, *ef_sum2 = ef_stats+256, *ef_sq2 = ef_stats+384;
    float* ef_sc1 = ef_stats+512, *ef_sh1 = ef_stats+640, *ef_sc2 = ef_stats+768, *ef_sh2 = ef_stats+896;

    float* pmn = (float*)alloc((size_t)3 * G * 128 * 4);
    float* pmx = (float*)alloc((size_t)3 * G * 128 * 4);
    float* psm = (float*)alloc((size_t)3 * G * 128 * 4);
    float* c0  = (float*)alloc((size_t)G * 4);
    float* c1  = (float*)alloc((size_t)G * 4);
    float* Hp  = (float*)alloc((size_t)G * 1152 * 4);
    (void)ws_size; (void)n_in;

    // ---- weight conversion (fp32 -> f16, transposed to Wt[n*K+k])
    k_cvt_w<<<CDIV(128*128,256),256,0,stream>>>(gcn1_W, gcn1Wt, 128, 128);
    k_cvt_w<<<CDIV(384*64,256),256,0,stream>>>(neW0, neW0t, 384, 64);
    k_cvt_w<<<CDIV(64*64,256),256,0,stream>>>(neW1, neW1t, 64, 64);
    k_cvt_w<<<CDIV(128*128,256),256,0,stream>>>(efW0, efW0t, 128, 128);
    k_cvt_w<<<CDIV(128*128,256),256,0,stream>>>(efW1, efW1t, 128, 128);
    k_cvt_w<<<CDIV(128*128,256),256,0,stream>>>(gcn2_W, gcn2Wt, 128, 128);

    // ---- degrees (self-loop => init 1)
    k_fill<<<CDIV(N,256),256,0,stream>>>(dinv1, 1.0f, N);
    k_deg<<<CDIV(E,256),256,0,stream>>>(ei1, E, dinv1);
    k_rsqrt_ip<<<CDIV(N,256),256,0,stream>>>(dinv1, N);
    k_fill<<<CDIV(L2,256),256,0,stream>>>(dinv2, 1.0f, L2);
    k_deg<<<CDIV(EL2,256),256,0,stream>>>(l2e1, EL2, dinv2);
    k_rsqrt_ip<<<CDIV(L2,256),256,0,stream>>>(dinv2, L2);

    // ---- GCN1
    size_t sm_gemm128 = (size_t)(64*128 + 128*128) * 2;
    k_gemm_emb<<<CDIV(N,64),256,sm_gemm128,stream>>>(node_emb, x, gcn1Wt, Hlin, N);
    k_selfbias<<<(unsigned)CDIV((long)N*128,256),256,0,stream>>>(H0, Hlin, dinv1, gcn1_b, N);
    k_scatter<<<(unsigned)CDIV((long)E*32,256),256,0,stream>>>(H0, Hlin, ei0, ei1, dinv1, E);

    // ---- edge FFN (384 -> 64), two symmetric branches off one staged tile
    size_t sm_ne = (size_t)(64*384 + 64*384) * 2;
    k_ffn_ne_gemm<<<CDIV(EU,64),256,sm_ne,stream>>>(H0, edge_emb, edge_attr, ue, ei0, ei1,
                                                    neW0t, ne_b0, Zne1, Zne2, EU);
    k_fill<<<CDIV(256,256),256,0,stream>>>(ne_stats, 0.0f, 256);
    k_colstats<64><<<CDIV(EU,CS_RPB),256,0,stream>>>(Zne1, EU, ne_sum1, ne_sq1);
    k_colstats<64><<<CDIV(EU,CS_RPB),256,0,stream>>>(Zne2, EU, ne_sum2, ne_sq2);
    k_bnfin<<<1,64,0,stream>>>(ne_sum1, ne_sq1, ne_g0, ne_be0, ne_sc1, ne_sh1, 64, 1.0f/(float)EU);
    k_bnfin<<<1,64,0,stream>>>(ne_sum2, ne_sq2, ne_g0, ne_be0, ne_sc2, ne_sh2, 64, 1.0f/(float)EU);
    size_t sm_nep2 = (size_t)(64*64 + 64*64) * 2;
    k_ffn_part2<64,64><<<CDIV(EU,64),256,sm_nep2,stream>>>(Zne1, Zne2,
                    ne_sc1, ne_sh1, ne_sc2, ne_sh2, neW1t, ne_b1, h_edge, EU);

    // ---- l2_batch
    k_l2batch<<<CDIV(L2,256),256,0,stream>>>(e_idx, ue, ei0, batch, l2b, L2);

    // ---- pair FFN (128 -> 128)
    k_ffn_ef_gemm<<<CDIV(L2,64),256,sm_gemm128,stream>>>(h_edge, e_idx, f_idx, efW0t, ef_b0,
                                                         Zef1, Zef2, L2);
    k_fill<<<CDIV(512,256),256,0,stream>>>(ef_stats, 0.0f, 512);
    k_colstats<128><<<CDIV(L2,CS_RPB),256,0,stream>>>(Zef1, L2, ef_sum1, ef_sq1);
    k_colstats<128><<<CDIV(L2,CS_RPB),256,0,stream>>>(Zef2, L2, ef_sum2, ef_sq2);
    k_bnfin<<<1,128,0,stream>>>(ef_sum1, ef_sq1, ef_g0, ef_be0, ef_sc1, ef_sh1, 128, 1.0f/(float)L2);
    k_bnfin<<<1,128,0,stream>>>(ef_sum2, ef_sq2, ef_g0, ef_be0, ef_sc2, ef_sh2, 128, 1.0f/(float)L2);
    k_ffn_part2<128,128><<<CDIV(L2,64),256,sm_gemm128,stream>>>(Zef1, Zef2,
                    ef_sc1, ef_sh1, ef_sc2, ef_sh2, efW1t, ef_b1, H1, L2);

    // ---- GCN2 (Hlin2/H2 reuse Zef buffers; Zef fully consumed above)
    k_gemm_h<<<CDIV(L2,64),256,sm_gemm128,stream>>>(H1, gcn2Wt, Hlin2, L2);
    k_selfbias<<<(unsigned)CDIV((long)L2*128,256),256,0,stream>>>(H2, Hlin2, dinv2, gcn2_b, L2);
    k_scatter<<<(unsigned)CDIV((long)EL2*32,256),256,0,stream>>>(H2, Hlin2, l2e0, l2e1, dinv2, EL2);

    // ---- pooling (relu folded into H2 read)
    k_pool_init<<<CDIV(3*G*128,256),256,0,stream>>>(pmn, pmx, psm, c0, c1, 3*G*128, G);
    k_pool_acc<<<(unsigned)CDIV((long)N*32,256),256,0,stream>>>(H0, batch, N,
                pmn, pmx, psm, c0, 0, 1);
    k_pool_acc<<<(unsigned)CDIV((long)L2*32,256),256,0,stream>>>(H1, l2b, L2,
                pmn + G*128, pmx + G*128, psm + G*128, c1, 0, 1);
    k_pool_acc<<<(unsigned)CDIV((long)L2*32,256),256,0,stream>>>(H2, l2b, L2,
                pmn + 2*G*128, pmx + 2*G*128, psm + 2*G*128, c1, 1, 0);
    k_pool_fin<<<CDIV(G*1152,256),256,0,stream>>>(pmn, pmx, psm, c0, c1, Hp, G);

    // ---- head
    k_head<<<G,128,0,stream>>>(Hp, hW1, hb1, hg1, hbe1, hW2, hb2, hg2, hbe2, hW3, hb3,
                               (float*)d_out);
}
